// HumanUVNerfMotion_88210038325496
// MI455X (gfx1250) — matrix operand order repaired and checked by reading, not compiled, not run on hardware
//
#include <hip/hip_runtime.h>
#include <hip/hip_bf16.h>

// ---------------------------------------------------------------------------
// HumanUVNerfMotion on MI455X (gfx1250, wave32, WMMA)
//   K1: nearest-neighbor (LDS-tiled, fp32 VALU)
//   K2: featurize -> fp16 WMMA A-fragments (embedding + bilinear plane fusion)
//   K3: weight pack -> fp16 WMMA B-fragments
//   K4: geo MLP  (128->256->256, +alpha head) all v_wmma_f32_16x16x32_f16
//   K5: view MLP (352->256->128, +rgb head)   all v_wmma_f32_16x16x32_f16
// ---------------------------------------------------------------------------

typedef __attribute__((ext_vector_type(16))) _Float16 v16h;
typedef __attribute__((ext_vector_type(8)))  float    v8f;

#define N_PTS   16384
#define NVERT   6890
#define RESO    256
#define PLANE   (RESO * RESO)
#define TILES   (N_PTS / 16)     // 1024 point-tiles of 16
#define GWAVES  4                // waves per GEMM block

static __device__ __forceinline__ v8f wmma16(const v16h a, const v16h b, v8f c) {
    return __builtin_amdgcn_wmma_f32_16x16x32_f16(false, a, false, b, (short)0, c, false, false);
}

// ------------------------------ K1: nearest neighbor ------------------------
#define VTILE 1024
__global__ __launch_bounds__(256) void nn_kernel(const float* __restrict__ pts,
                                                 const float* __restrict__ verts,
                                                 int* __restrict__ outIdx,
                                                 float* __restrict__ outH) {
    __shared__ float sv[VTILE * 3];
    const int n = blockIdx.x * 256 + threadIdx.x;
    const float px = pts[n * 3 + 0], py = pts[n * 3 + 1], pz = pts[n * 3 + 2];
    float best = 3.0e38f;
    int bestI = 0;
    for (int base = 0; base < NVERT; base += VTILE) {
        const int cnt = min(VTILE, NVERT - base);
        __syncthreads();
        for (int i = threadIdx.x; i < cnt * 3; i += 256) sv[i] = verts[base * 3 + i];
        __syncthreads();
        for (int i = 0; i < cnt; ++i) {
            const float dx = px - sv[i * 3 + 0];
            const float dy = py - sv[i * 3 + 1];
            const float dz = pz - sv[i * 3 + 2];
            const float d = fmaf(dx, dx, fmaf(dy, dy, dz * dz));
            if (d < best) { best = d; bestI = base + i; }
        }
    }
    outIdx[n] = bestI;
    outH[n] = sqrtf(fmaxf(best, 1e-12f));
}

// ------------------------------ K2: featurize -------------------------------
__device__ __forceinline__ void bilin_acc(const float* __restrict__ pl,
                                          float uc, float vc, float* acc) {
    const float x = fminf(fmaxf(uc, 0.f), 1.f) * (float)(RESO - 1);
    const float y = fminf(fmaxf(vc, 0.f), 1.f) * (float)(RESO - 1);
    const float x0f = floorf(x), y0f = floorf(y);
    const float fx = x - x0f, fy = y - y0f;
    int x0 = (int)x0f; x0 = max(0, min(RESO - 1, x0));
    int y0 = (int)y0f; y0 = max(0, min(RESO - 1, y0));
    const int x1 = min(x0 + 1, RESO - 1);
    const int y1 = min(y0 + 1, RESO - 1);
    const float w00 = (1.f - fx) * (1.f - fy), w01 = fx * (1.f - fy);
    const float w10 = (1.f - fx) * fy,         w11 = fx * fy;
    const int i00 = y0 * RESO + x0, i01 = y0 * RESO + x1;
    const int i10 = y1 * RESO + x0, i11 = y1 * RESO + x1;
#pragma unroll
    for (int c = 0; c < 64; ++c) {
        const float* p = pl + (size_t)c * PLANE;
        acc[c] += p[i00] * w00 + p[i01] * w01 + p[i10] * w10 + p[i11] * w11;
    }
}

__global__ __launch_bounds__(256) void featurize_kernel(const float* __restrict__ viewdir,
                                                        const float* __restrict__ latent,
                                                        const float* __restrict__ vtsuv,
                                                        const int* __restrict__ nnIdx,
                                                        const float* __restrict__ nnH,
                                                        _Float16* __restrict__ x0frag,
                                                        _Float16* __restrict__ xvfrag) {
    const int n = blockIdx.x * 256 + threadIdx.x;
    const int idx = nnIdx[n];
    const float h = nnH[n];
    const float u = vtsuv[idx * 2 + 0];
    const float v = vtsuv[idx * 2 + 1];
    const float hn = (h * 10.0f + 1.0f) * 0.5f;   // (h/DEPTH_BOUND + 1)/2

    float F[128];
#pragma unroll
    for (int c = 0; c < 64; ++c) F[c] = 0.f;
    bilin_acc(latent,                      u,  v,  F);   // uv plane
    bilin_acc(latent + (size_t)64 * PLANE, u,  hn, F);   // uh plane
    bilin_acc(latent + (size_t)128 * PLANE, hn, v, F);   // hv plane
#pragma unroll
    for (int c = 0; c < 64; ++c) F[c] *= (1.f / 3.f);

    // positional embedding: [h,u,v, sin(3x10), cos(3x10)]
    F[64] = h; F[65] = u; F[66] = v;
    float fr = 1.f;
#pragma unroll
    for (int f = 0; f < 10; ++f) {
        F[67 + f]  = __sinf(h * fr); F[77 + f]  = __sinf(u * fr); F[87 + f]  = __sinf(v * fr);
        F[97 + f]  = __cosf(h * fr); F[107 + f] = __cosf(u * fr); F[117 + f] = __cosf(v * fr);
        fr *= 2.f;
    }
    F[127] = 0.f;

    const int t = n >> 4, m = n & 15;
    // A-fragment layout (16-bit 16x32): lane<16 holds K {0..7,16..23}, lane>=16 {8..15,24..31}
#pragma unroll
    for (int kt = 0; kt < 4; ++kt) {
#pragma unroll
        for (int j = 0; j < 16; ++j) {
            const int ko = (j < 8) ? j : j + 8;
            x0frag[(((size_t)t * 4 + kt) * 32 + m)      * 16 + j] = (_Float16)F[kt * 32 + ko];
            x0frag[(((size_t)t * 4 + kt) * 32 + m + 16) * 16 + j] = (_Float16)F[kt * 32 + 8 + ko];
        }
    }
    // view-stage tail features (K 256..351 of X3): view(27) | fused(64) | pad(5)
#pragma unroll
    for (int kt = 0; kt < 3; ++kt) {
#pragma unroll
        for (int j = 0; j < 16; ++j) {
            const int ko = (j < 8) ? j : j + 8;
            const int k0 = kt * 32 + ko, k1 = kt * 32 + 8 + ko;
            const float g0 = (k0 < 27) ? viewdir[(size_t)n * 27 + k0] : (k0 < 91 ? F[k0 - 27] : 0.f);
            const float g1 = (k1 < 27) ? viewdir[(size_t)n * 27 + k1] : (k1 < 91 ? F[k1 - 27] : 0.f);
            xvfrag[(((size_t)t * 3 + kt) * 32 + m)      * 16 + j] = (_Float16)g0;
            xvfrag[(((size_t)t * 3 + kt) * 32 + m + 16) * 16 + j] = (_Float16)g1;
        }
    }
}

// ------------------------------ K3: weight pack -----------------------------
// B-fragment (16-bit 32x16): lanes 0-15 hold K {0..7,16..23} of the k-tile, lanes 16-31
// hold K {8..15,24..31}. N = ntile*16 + lane%16.
__global__ void pack_w_kernel(const float* __restrict__ W, _Float16* __restrict__ dst,
                              int Kin, int Nout, int ntiles, int total) {
    const int e = blockIdx.x * blockDim.x + threadIdx.x;
    if (e >= total) return;
    const int j = e & 15;
    const int lane = (e >> 4) & 31;
    const int fi = e >> 9;
    const int kt = fi / ntiles, nt = fi % ntiles;
    const int k = kt * 32 + ((lane & 16) ? 8 : 0) + ((j < 8) ? j : j + 8);
    const int nn = nt * 16 + (lane & 15);
    const float v = (k < Kin && nn < Nout) ? W[(size_t)nn * Kin + k] : 0.f;
    dst[e] = (_Float16)v;
}

// ----------------------- A-fragment rebuild from LDS ------------------------
__device__ __forceinline__ v16h frag_from_stage(const unsigned* __restrict__ rowW,
                                                int kt, int mhi) {
    v16h a;
    unsigned* ap = (unsigned*)&a;
    const int b0 = (kt * 32 + mhi) >> 1;   // mhi = 0 or 8 (K half select)
#pragma unroll
    for (int j = 0; j < 4; ++j) ap[j] = rowW[b0 + j];
#pragma unroll
    for (int j = 0; j < 4; ++j) ap[4 + j] = rowW[b0 + 8 + j];
    return a;
}

// ------------------------------ K4: geo MLP ---------------------------------
__global__ __launch_bounds__(GWAVES * 32) void geo_kernel(
        const _Float16* __restrict__ x0frag, const _Float16* __restrict__ w0f,
        const _Float16* __restrict__ w1f,    const _Float16* __restrict__ waf,
        const float* __restrict__ b0, const float* __restrict__ b1,
        const float* __restrict__ bA,
        _Float16* __restrict__ y2frag, float* __restrict__ out) {
    __shared__ _Float16 stage[GWAVES][16][256];
    const int lane = threadIdx.x & 31;
    const int wave = threadIdx.x >> 5;
    const int tile = blockIdx.x * GWAVES + wave;
    const int nlo = lane & 15;
    const int mhi = (lane >> 4) << 3;   // 0 or 8
    const unsigned* rowW = (const unsigned*)(&stage[wave][nlo][0]);

    v16h a0[4];
#pragma unroll
    for (int kt = 0; kt < 4; ++kt)
        a0[kt] = *(const v16h*)(x0frag + (((size_t)tile * 4 + kt) * 32 + lane) * 16);

    // ---- geo0: 128 -> 256, relu
#pragma unroll 1
    for (int nt = 0; nt < 16; ++nt) {
        if (nt + 1 < 16)
            __builtin_prefetch(w0f + (((size_t)(nt + 1)) * 32 + lane) * 16, 0, 0);
        v8f acc = {};
#pragma unroll
        for (int kt = 0; kt < 4; ++kt) {
            v16h b = *(const v16h*)(w0f + (((size_t)kt * 16 + nt) * 32 + lane) * 16);
            acc = wmma16(a0[kt], b, acc);
        }
        const float bias = b0[nt * 16 + nlo];
#pragma unroll
        for (int r = 0; r < 8; ++r) {
            float v = acc[r] + bias; v = v > 0.f ? v : 0.f;
            stage[wave][r + mhi][nt * 16 + nlo] = (_Float16)v;
        }
    }
    __syncthreads();

    v16h a1[8];
#pragma unroll
    for (int kt = 0; kt < 8; ++kt) a1[kt] = frag_from_stage(rowW, kt, mhi);
    __syncthreads();

    // ---- geo1: 256 -> 256, relu
#pragma unroll 1
    for (int nt = 0; nt < 16; ++nt) {
        v8f acc = {};
#pragma unroll
        for (int kt = 0; kt < 8; ++kt) {
            v16h b = *(const v16h*)(w1f + (((size_t)kt * 16 + nt) * 32 + lane) * 16);
            acc = wmma16(a1[kt], b, acc);
        }
        const float bias = b1[nt * 16 + nlo];
#pragma unroll
        for (int r = 0; r < 8; ++r) {
            float v = acc[r] + bias; v = v > 0.f ? v : 0.f;
            stage[wave][r + mhi][nt * 16 + nlo] = (_Float16)v;
        }
    }
    __syncthreads();

    // ---- export Y2 fragments + alpha head (padded to one 16-wide N-tile)
    v8f accA = {};
#pragma unroll
    for (int kt = 0; kt < 8; ++kt) {
        v16h a = frag_from_stage(rowW, kt, mhi);
        *(v16h*)(y2frag + (((size_t)tile * 8 + kt) * 32 + lane) * 16) = a;
        v16h b = *(const v16h*)(waf + ((size_t)kt * 32 + lane) * 16);
        accA = wmma16(a, b, accA);
    }
    if (nlo == 0) {
        const float bias = bA[0];
#pragma unroll
        for (int r = 0; r < 8; ++r)
            out[((size_t)tile * 16 + r + mhi) * 4 + 0] = accA[r] + bias;
    }
}

// ------------------------------ K5: view MLP --------------------------------
__global__ __launch_bounds__(GWAVES * 32) void view_kernel(
        const _Float16* __restrict__ y2frag, const _Float16* __restrict__ xvfrag,
        const _Float16* __restrict__ w2f, const _Float16* __restrict__ w3f,
        const _Float16* __restrict__ wrf,
        const float* __restrict__ bv0, const float* __restrict__ bv1,
        const float* __restrict__ brgb, float* __restrict__ out) {
    __shared__ _Float16 stage[GWAVES][16][256];
    const int lane = threadIdx.x & 31;
    const int wave = threadIdx.x >> 5;
    const int tile = blockIdx.x * GWAVES + wave;
    const int nlo = lane & 15;
    const int mhi = (lane >> 4) << 3;
    const unsigned* rowW = (const unsigned*)(&stage[wave][nlo][0]);

    v16h a[11];
#pragma unroll
    for (int kt = 0; kt < 8; ++kt)
        a[kt] = *(const v16h*)(y2frag + (((size_t)tile * 8 + kt) * 32 + lane) * 16);
#pragma unroll
    for (int kt = 0; kt < 3; ++kt)
        a[8 + kt] = *(const v16h*)(xvfrag + (((size_t)tile * 3 + kt) * 32 + lane) * 16);

    // ---- view0: 352 -> 256, relu
#pragma unroll 1
    for (int nt = 0; nt < 16; ++nt) {
        if (nt + 1 < 16)
            __builtin_prefetch(w2f + (((size_t)(nt + 1)) * 32 + lane) * 16, 0, 0);
        v8f acc = {};
#pragma unroll
        for (int kt = 0; kt < 11; ++kt) {
            v16h b = *(const v16h*)(w2f + (((size_t)kt * 16 + nt) * 32 + lane) * 16);
            acc = wmma16(a[kt], b, acc);
        }
        const float bias = bv0[nt * 16 + nlo];
#pragma unroll
        for (int r = 0; r < 8; ++r) {
            float v = acc[r] + bias; v = v > 0.f ? v : 0.f;
            stage[wave][r + mhi][nt * 16 + nlo] = (_Float16)v;
        }
    }
    __syncthreads();

    v16h a4[8];
#pragma unroll
    for (int kt = 0; kt < 8; ++kt) a4[kt] = frag_from_stage(rowW, kt, mhi);
    __syncthreads();

    // ---- view1: 256 -> 128, relu (cols 0..127 of stage)
#pragma unroll 1
    for (int nt = 0; nt < 8; ++nt) {
        v8f acc = {};
#pragma unroll
        for (int kt = 0; kt < 8; ++kt) {
            v16h b = *(const v16h*)(w3f + (((size_t)kt * 8 + nt) * 32 + lane) * 16);
            acc = wmma16(a4[kt], b, acc);
        }
        const float bias = bv1[nt * 16 + nlo];
#pragma unroll
        for (int r = 0; r < 8; ++r) {
            float v = acc[r] + bias; v = v > 0.f ? v : 0.f;
            stage[wave][r + mhi][nt * 16 + nlo] = (_Float16)v;
        }
    }
    __syncthreads();

    // ---- rgb head: 128 -> 3 (padded to one 16-wide N-tile)
    v8f accR = {};
#pragma unroll
    for (int kt = 0; kt < 4; ++kt) {
        v16h af = frag_from_stage(rowW, kt, mhi);
        v16h b = *(const v16h*)(wrf + ((size_t)kt * 32 + lane) * 16);
        accR = wmma16(af, b, accR);
    }
    if (nlo < 3) {
        const float bias = brgb[nlo];
#pragma unroll
        for (int r = 0; r < 8; ++r)
            out[((size_t)tile * 16 + r + mhi) * 4 + 1 + nlo] = accR[r] + bias;
    }
}

// ------------------------------ launch --------------------------------------
extern "C" void kernel_launch(void* const* d_in, const int* in_sizes, int n_in,
                              void* d_out, int out_size, void* d_ws, size_t ws_size,
                              hipStream_t stream) {
    const float* pts     = (const float*)d_in[0];
    const float* verts   = (const float*)d_in[1];
    const float* viewdir = (const float*)d_in[2];
    const float* latent  = (const float*)d_in[3];
    const float* vtsuv   = (const float*)d_in[4];
    const float* w_geo0  = (const float*)d_in[5];
    const float* b_geo0  = (const float*)d_in[6];
    const float* w_geo1  = (const float*)d_in[7];
    const float* b_geo1  = (const float*)d_in[8];
    const float* w_alpha = (const float*)d_in[9];
    const float* b_alpha = (const float*)d_in[10];
    const float* w_view0 = (const float*)d_in[11];
    const float* b_view0 = (const float*)d_in[12];
    const float* w_view1 = (const float*)d_in[13];
    const float* b_view1 = (const float*)d_in[14];
    const float* w_rgb   = (const float*)d_in[15];
    const float* b_rgb   = (const float*)d_in[16];
    float* out = (float*)d_out;

    char* ws = (char*)d_ws;
    size_t off = 0;
    auto carve = [&](size_t bytes) -> char* {
        char* p = ws + off;
        off = (off + bytes + 255) & ~(size_t)255;
        return p;
    };
    int*      nnIdx  = (int*)     carve((size_t)N_PTS * 4);
    float*    nnH    = (float*)   carve((size_t)N_PTS * 4);
    _Float16* x0f    = (_Float16*)carve((size_t)TILES * 4 * 512 * 2);
    _Float16* xvf    = (_Float16*)carve((size_t)TILES * 3 * 512 * 2);
    _Float16* y2f    = (_Float16*)carve((size_t)TILES * 8 * 512 * 2);
    _Float16* w0f    = (_Float16*)carve((size_t)4 * 16 * 512 * 2);
    _Float16* w1f    = (_Float16*)carve((size_t)8 * 16 * 512 * 2);
    _Float16* waf    = (_Float16*)carve((size_t)8 * 1 * 512 * 2);
    _Float16* w2f    = (_Float16*)carve((size_t)11 * 16 * 512 * 2);
    _Float16* w3f    = (_Float16*)carve((size_t)8 * 8 * 512 * 2);
    _Float16* wrf    = (_Float16*)carve((size_t)4 * 1 * 512 * 2);

    // weight packing (L2-resident, done per call; cheap)
    pack_w_kernel<<<(32768 + 255) / 256, 256, 0, stream>>>(w_geo0,  w0f, 127, 256, 16, 32768);
    pack_w_kernel<<<(65536 + 255) / 256, 256, 0, stream>>>(w_geo1,  w1f, 256, 256, 16, 65536);
    pack_w_kernel<<<(4096  + 255) / 256, 256, 0, stream>>>(w_alpha, waf, 256, 1,   1,  4096);
    pack_w_kernel<<<(90112 + 255) / 256, 256, 0, stream>>>(w_view0, w2f, 347, 256, 16, 90112);
    pack_w_kernel<<<(32768 + 255) / 256, 256, 0, stream>>>(w_view1, w3f, 256, 128, 8,  32768);
    pack_w_kernel<<<(2048  + 255) / 256, 256, 0, stream>>>(w_rgb,   wrf, 128, 3,   1,  2048);

    nn_kernel<<<N_PTS / 256, 256, 0, stream>>>(pts, verts, nnIdx, nnH);
    featurize_kernel<<<N_PTS / 256, 256, 0, stream>>>(viewdir, latent, vtsuv, nnIdx, nnH, x0f, xvf);
    geo_kernel<<<TILES / GWAVES, GWAVES * 32, 0, stream>>>(x0f, w0f, w1f, waf,
                                                           b_geo0, b_geo1, b_alpha, y2f, out);
    view_kernel<<<TILES / GWAVES, GWAVES * 32, 0, stream>>>(y2f, xvf, w2f, w3f, wrf,
                                                            b_view0, b_view1, b_rgb, out);
}
